// biLSTMModelSelfAttention_22454089023695
// MI455X (gfx1250) — matrix operand order repaired and checked by reading, not compiled, not run on hardware
//
#include <hip/hip_runtime.h>

#define B_  32
#define T_  1024
#define IN_ 256
#define H_  128
#define F2_ 256
#define G4_ 512   // 4*H
#define NC_ 4

typedef __attribute__((ext_vector_type(16))) _Float16 v16h;
typedef __attribute__((ext_vector_type(8)))  float    v8f;

// ---------------- WMMA fragment helpers (wave32, v_wmma_f32_16x16x32_f16) ----
// A fragment: 16(M) x 32(K), source row-major [m][k], leading dim lda (halfs).
// Contiguous-k layout (same as frag_b): lanes 0-15 -> M=lane, elems = K0..15;
//             lanes 16-31 -> M=lane-16, elems = K16..31.
__device__ inline v16h frag_a(const _Float16* src, int lda, int lane) {
  const int m  = lane & 15;
  const int kb = (lane >> 4) << 4;            // 0 or 16
  const _Float16* p = src + m * lda;
  v16h a;
#pragma unroll
  for (int i = 0; i < 8; ++i) a[i] = p[kb + i];
#pragma unroll
  for (int i = 0; i < 8; ++i) a[8 + i] = p[kb + 8 + i];
  return a;
}

// B fragment: 32(K) x 16(N), source stored n-major: src[n*ldb + k].
// Contiguous-k layout (same as frag_a): lanes 0-15 -> N=lane, elems = K0..15; lanes 16-31 -> N=lane-16, K16..31.
__device__ inline v16h frag_b(const _Float16* src, int ldb, int lane) {
  const int n  = lane & 15;
  const int kb = (lane >> 4) << 4;            // 0 or 16
  const _Float16* p = src + n * ldb + kb;
  v16h b;
#pragma unroll
  for (int i = 0; i < 16; ++i) b[i] = p[i];
  return b;
}

// ---------------- Generic f32-in / f16-WMMA GEMM -----------------------------
// C[m][n] = sum_k A[m][k] * Bt[n][k] + bias0[n] (+ bias1[n])
// Tile: 64(M) x 128(N) per 256-thread block (8 waves, 4 subtiles each).
// permuteTB=1 stores row m=b*T+t to row t*B+b; outF16=1 stores _Float16.
__global__ void __launch_bounds__(256)
gemm_f16_wmma(const float* __restrict__ A, long rowStrideA,
              const float* __restrict__ Bt,
              const float* __restrict__ bias0,
              const float* __restrict__ bias1,
              void* __restrict__ C,
              int M, int N, int K, int permuteTB, int outF16)
{
  __shared__ _Float16 As[64][40];    // 64 x 32 (+8 pad)
  __shared__ _Float16 Bs[128][40];   // n-major: 128 n x 32 k (+8 pad)

  const int tid  = threadIdx.x;
  const int lane = tid & 31;
  const int w    = tid >> 5;
  const int mi   = (w & 3) * 16;
  const int nb   = (w >> 2) * 64;

  const int m0 = blockIdx.x * 64;
  const int n0 = blockIdx.y * 128;

  v8f acc[4] = {};

  const int arow = tid >> 2;              // 0..63
  const int acol = (tid & 3) * 8;
  const int brow = tid >> 1;              // 0..127
  const int bcol = (tid & 1) * 16;

  for (int k0 = 0; k0 < K; k0 += 32) {
    if (m0 + arow < M) {
      const float* ap = A + (long)(m0 + arow) * rowStrideA + k0 + acol;
#pragma unroll
      for (int i = 0; i < 8; ++i) As[arow][acol + i] = (_Float16)ap[i];
    } else {
#pragma unroll
      for (int i = 0; i < 8; ++i) As[arow][acol + i] = (_Float16)0.f;
    }
    {
      const float* bp = Bt + (long)(n0 + brow) * K + k0 + bcol;
#pragma unroll
      for (int i = 0; i < 16; ++i) Bs[brow][bcol + i] = (_Float16)bp[i];
    }
    __syncthreads();

    // batch fragment loads, then issue WMMAs back-to-back
    v16h a = frag_a(&As[mi][0], 40, lane);
    v16h bf[4];
#pragma unroll
    for (int s = 0; s < 4; ++s) bf[s] = frag_b(&Bs[nb + s * 16][0], 40, lane);
#pragma unroll
    for (int s = 0; s < 4; ++s)
      acc[s] = __builtin_amdgcn_wmma_f32_16x16x32_f16(
                   false, a, false, bf[s], (short)0, acc[s], false, false);
    __syncthreads();
  }

  // C/D layout: lanes 0-15 -> N=lane, M=r; lanes 16-31 -> N=lane-16, M=r+8.
  const int nloc = lane & 15;
  const int mloc = (lane >> 4) * 8;
#pragma unroll
  for (int s = 0; s < 4; ++s) {
    const int ng = n0 + nb + s * 16 + nloc;
    const float bvv = bias0[ng] + (bias1 ? bias1[ng] : 0.f);
#pragma unroll
    for (int r = 0; r < 8; ++r) {
      const int mg = m0 + mi + mloc + r;
      if (mg < M) {
        const long row = permuteTB ? (long)((mg & (T_ - 1)) * B_ + (mg >> 10))
                                   : (long)mg;
        const float v = acc[s][r] + bvv;
        if (outF16) ((_Float16*)C)[row * N + ng] = (_Float16)v;
        else        ((float*)C)[row * N + ng] = v;
      }
    }
  }
}

// ---------------- async prefetch of one timestep's gate slab -----------------
// gx is (T, B, 1024) f16. One direction's slab = 32 rows x 512 f16 (1KB/row,
// row stride 2KB). 512 threads x 4 x 16B async-copies = 32KB.
// LDS rows padded to 1040B (bank-spread) -> 33,280B per buffer.
__device__ inline void gx_prefetch(const _Float16* gx, int ta, int dir,
                                   _Float16* dstbuf, int tid) {
  const char* gbase = (const char*)(gx + ((size_t)ta * B_) * 1024 + dir * 512);
  const unsigned lbase = (unsigned)(unsigned long long)(const void*)dstbuf;
#pragma unroll
  for (int i = 0; i < 4; ++i) {
    const int q = tid + 512 * i;         // 0..2047 16B-chunks
    const int m = q >> 6;                // row (batch)
    const int j = q & 63;                // 16B chunk within row
    const unsigned ldsoff = lbase + (unsigned)(m * 1040 + j * 16);
    const unsigned long long ga =
        (unsigned long long)(gbase + (size_t)m * 2048 + (size_t)j * 16);
    asm volatile("global_load_async_to_lds_b128 %0, %1, off"
                 :: "v"(ldsoff), "v"(ga) : "memory");
  }
}

// ---------------- Persistent LSTM recurrence (one block per direction) ------
// W_hh (512x128 f16, n-major) resident in LDS; gx double-buffered into LDS by
// CDNA5 async loads overlapping each step's WMMA block. Per step:
//   [issue async slab t+1] -> WMMA g=h@Whh^T -> gs -> s_wait_asynccnt/barrier
//   -> gate math (gs + gx slab t) -> h (f16 LDS) / out -> barrier.
__global__ void __launch_bounds__(512)
lstm_rec_wmma(const _Float16* __restrict__ gx,  // (T, B, 1024) incl. biases
              const float* __restrict__ whh,    // (2, 512, 128) this layer
              float* __restrict__ out)          // (B, T, 256)
{
  extern __shared__ char smem[];
  _Float16* whhs = (_Float16*)smem;                 // [512][136]
  _Float16* hs   = whhs + 512 * 136;                // [32][136]
  float*    gs   = (float*)(hs + 32 * 136);         // [32][516]
  _Float16* gxs  = (_Float16*)(gs + 32 * 516);      // 2 x [32][520]

  const int dir  = blockIdx.x;
  const int tid  = threadIdx.x;
  const int lane = tid & 31;
  const int w    = tid >> 5;                        // 0..15
  const int mi   = (w & 1) * 16;
  const int nbw  = (w >> 1) * 64;

  // kick off slab for t=0 immediately; it overlaps the 139KB weight staging
  gx_prefetch(gx, dir ? (T_ - 1) : 0, dir, gxs, tid);

  // stage W_hh n-major as f16
  {
    const float* wp = whh + (long)dir * G4_ * H_ + (long)tid * H_;
    _Float16* dp = whhs + tid * 136;
#pragma unroll 8
    for (int k = 0; k < H_; ++k) dp[k] = (_Float16)wp[k];
  }
  for (int i = tid; i < 32 * 136; i += 512) hs[i] = (_Float16)0.f;

  float c[8];
#pragma unroll
  for (int r = 0; r < 8; ++r) c[r] = 0.f;

  __syncthreads();

  const int nloc = lane & 15;
  const int mloc = (lane >> 4) * 8;

  for (int t = 0; t < T_; ++t) {
    const int ta = dir ? (T_ - 1 - t) : t;

    // prefetch slab t+1 into the other buffer (read of that buffer at t-1
    // finished before the barrier that ended the previous iteration)
    if (t + 1 < T_)
      gx_prefetch(gx, dir ? (T_ - 2 - t) : (t + 1), dir,
                  gxs + ((t + 1) & 1) * (32 * 520), tid);

    v8f acc[4] = {};
#pragma unroll
    for (int kk = 0; kk < 4; ++kk) {
      v16h a = frag_a(hs + mi * 136 + kk * 32, 136, lane);
      v16h bf[4];
#pragma unroll
      for (int s = 0; s < 4; ++s)
        bf[s] = frag_b(whhs + (nbw + s * 16) * 136 + kk * 32, 136, lane);
#pragma unroll
      for (int s = 0; s < 4; ++s)
        acc[s] = __builtin_amdgcn_wmma_f32_16x16x32_f16(
                     false, a, false, bf[s], (short)0, acc[s], false, false);
    }

    // raw h@Whh^T partials to LDS
#pragma unroll
    for (int s = 0; s < 4; ++s) {
      const int n = nbw + s * 16 + nloc;
#pragma unroll
      for (int r = 0; r < 8; ++r)
        gs[(mi + mloc + r) * 516 + n] = acc[s][r];
    }

    // slab t complete when ASYNCcnt drops to the 4 in-flight for slab t+1
    if (t + 1 < T_) asm volatile("s_wait_asynccnt 4" ::: "memory");
    else            asm volatile("s_wait_asynccnt 0" ::: "memory");
    __syncthreads();

    const _Float16* gxl = gxs + (t & 1) * (32 * 520);
#pragma unroll
    for (int r = 0; r < 8; ++r) {
      const int idx = r * 512 + tid;
      const int m = idx >> 7;
      const int j = idx & 127;
      const float* g = gs + m * 516;
      const _Float16* gm = gxl + m * 520;
      const float ig = 1.f / (1.f + expf(-(g[j] + (float)gm[j])));
      const float fg = 1.f / (1.f + expf(-(g[128 + j] + (float)gm[128 + j])));
      const float gg = tanhf(g[256 + j] + (float)gm[256 + j]);
      const float og = 1.f / (1.f + expf(-(g[384 + j] + (float)gm[384 + j])));
      c[r] = fg * c[r] + ig * gg;
      const float h = og * tanhf(c[r]);
      hs[m * 136 + j] = (_Float16)h;
      out[((long)m * T_ + ta) * F2_ + dir * H_ + j] = h;
    }
    __syncthreads();
  }
}

// ---------------- Attention epilogue: only t = T-1 row of ctx matters -------
__global__ void __launch_bounds__(256)
attn_last(const float* __restrict__ q,   // (B,256) with bias
          const float* __restrict__ Kb,  // (B,T,256)
          const float* __restrict__ Vb,  // (B,T,256)
          const float* __restrict__ wfc, // (4,256)
          const float* __restrict__ bfc,
          float* __restrict__ outp)      // (B,4)
{
  __shared__ float qs[256];
  __shared__ float sc[T_];
  __shared__ float red[256];
  __shared__ float ctx[256];

  const int b = blockIdx.x;
  const int tid = threadIdx.x;

  qs[tid] = q[b * 256 + tid];
  __syncthreads();

  const float scale = 0.0625f; // 1/sqrt(256)
  for (int s = tid; s < T_; s += 256) {
    const float* kr = Kb + ((long)b * T_ + s) * 256;
    float d = 0.f;
    for (int j = 0; j < 256; ++j) d += qs[j] * kr[j];
    sc[s] = d * scale;
  }
  __syncthreads();

  float mx = -1e30f;
  for (int s = tid; s < T_; s += 256) mx = fmaxf(mx, sc[s]);
  red[tid] = mx; __syncthreads();
  for (int o = 128; o > 0; o >>= 1) {
    if (tid < o) red[tid] = fmaxf(red[tid], red[tid + o]);
    __syncthreads();
  }
  mx = red[0]; __syncthreads();

  float sm = 0.f;
  for (int s = tid; s < T_; s += 256) {
    const float e = expf(sc[s] - mx);
    sc[s] = e; sm += e;
  }
  red[tid] = sm; __syncthreads();
  for (int o = 128; o > 0; o >>= 1) {
    if (tid < o) red[tid] += red[tid + o];
    __syncthreads();
  }
  const float inv = 1.f / red[0];
  __syncthreads();

  float a = 0.f;
  const float* vb = Vb + (long)b * T_ * 256 + tid;
  for (int s = 0; s < T_; ++s) a += sc[s] * vb[(long)s * 256];
  ctx[tid] = a * inv;
  __syncthreads();

  if (tid < NC_) {
    float o = bfc[tid];
    const float* wr = wfc + tid * 256;
    for (int j = 0; j < 256; ++j) o += ctx[j] * wr[j];
    outp[b * NC_ + tid] = o;
  }
}

// ---------------- host launcher ---------------------------------------------
extern "C" void kernel_launch(void* const* d_in, const int* in_sizes, int n_in,
                              void* d_out, int out_size, void* d_ws, size_t ws_size,
                              hipStream_t stream)
{
  (void)in_sizes; (void)n_in; (void)out_size; (void)ws_size;
  const float* x    = (const float*)d_in[0];
  const float* wih  = (const float*)d_in[1];
  const float* whh  = (const float*)d_in[2];
  const float* bih  = (const float*)d_in[3];
  const float* bhh  = (const float*)d_in[4];
  const float* wq   = (const float*)d_in[5];
  const float* wk   = (const float*)d_in[6];
  const float* wv   = (const float*)d_in[7];
  const float* bq   = (const float*)d_in[8];
  const float* bk   = (const float*)d_in[9];
  const float* bv   = (const float*)d_in[10];
  const float* wfc  = (const float*)d_in[11];
  const float* bfc  = (const float*)d_in[12];
  float* out = (float*)d_out;

  char* wsb = (char*)d_ws;
  _Float16* gx = (_Float16*)wsb;                          // (T,B,1024) f16
  float* out0 = (float*)(wsb + (size_t)T_ * B_ * 1024 * sizeof(_Float16));
  float* out1 = out0 + (size_t)B_ * T_ * F2_;
  float* kbuf = out1 + (size_t)B_ * T_ * F2_;
  float* vbuf = kbuf + (size_t)B_ * T_ * F2_;
  float* qbuf = vbuf + (size_t)B_ * T_ * F2_;             // B*256

  const size_t ldsRec = (size_t)(512 * 136 + 32 * 136) * sizeof(_Float16)
                      + (size_t)(32 * 516) * sizeof(float)
                      + (size_t)(2 * 32 * 520) * sizeof(_Float16);  // 280,576B

  const int MT = B_ * T_;
  dim3 blk(256);

  for (int l = 0; l < 2; ++l) {
    const float* in = (l == 0) ? x : out0;
    float* o = (l == 0) ? out0 : out1;
    gemm_f16_wmma<<<dim3(MT / 64, 1024 / 128), blk, 0, stream>>>(
        in, 256, wih + (size_t)l * 2 * G4_ * IN_,
        bih + (size_t)l * 2 * G4_, bhh + (size_t)l * 2 * G4_,
        (void*)gx, MT, 1024, 256, 1, 1);
    lstm_rec_wmma<<<dim3(2), dim3(512), ldsRec, stream>>>(
        gx, whh + (size_t)l * 2 * G4_ * H_, o);
  }

  gemm_f16_wmma<<<dim3(MT / 64, 2), blk, 0, stream>>>(
      out1, 256, wk, bk, nullptr, (void*)kbuf, MT, 256, 256, 0, 0);
  gemm_f16_wmma<<<dim3(MT / 64, 2), blk, 0, stream>>>(
      out1, 256, wv, bv, nullptr, (void*)vbuf, MT, 256, 256, 0, 0);
  gemm_f16_wmma<<<dim3(1, 2), blk, 0, stream>>>(
      out1 + (size_t)(T_ - 1) * F2_, (long)T_ * F2_,
      wq, bq, nullptr, (void*)qbuf, B_, 256, 256, 0, 0);

  attn_last<<<dim3(B_), blk, 0, stream>>>(qbuf, kbuf, vbuf, wfc, bfc, out);
}
